// BatchRelationalModule_10917806867210
// MI455X (gfx1250) — matrix operand order repaired and checked by reading, not compiled, not run on hardware
//
#include <hip/hip_runtime.h>
#include <hip/hip_bf16.h>

// ---------------------------------------------------------------------------
// Relational network for MI455X (gfx1250, wave32, WMMA).
//
// Math: out[b] = head( sum_{p,q} g(concat(feats_q, feats_p)) )
// Layer-0 of g is linear in the concat, so it collapses to
//   h0 = relu(A[q] + B[p])  with A = feats·Wg0[:, :65]^T + bg0,
//                                 B = feats·Wg0[:, 65:]^T
// leaving only two 64x64 matmuls per pair -> v_wmma_f32_16x16x32_f16.
//
// v2: weight fragments staged in LDS instead of registers to keep the wave
// under 256 VGPRs (kills s_set_vgpr_msb thrash, restores occupancy) and to
// give the scheduler independent DS work between WMMA issue and relu use.
// ---------------------------------------------------------------------------

typedef __attribute__((ext_vector_type(16))) _Float16 v16h;
typedef __attribute__((ext_vector_type(8)))  _Float16 v8h;
typedef __attribute__((ext_vector_type(8)))  float    v8f;
typedef __attribute__((ext_vector_type(4)))  float    v4f;

#define STRIDE 72   // halves per staged row; 72*2=144 bytes, multiple of 16

// Wave-internal LDS ordering: staging buffers are wave-private, LDS ops are
// in-order per wave, so a dscnt drain makes lane-crossing data visible.
__device__ __forceinline__ void lds_fence() {
  __builtin_amdgcn_wave_barrier();
  asm volatile("s_wait_dscnt 0" ::: "memory");
  __builtin_amdgcn_wave_barrier();
}

__device__ __forceinline__ v16h cat8(v8h lo, v8h hi) {
  return __builtin_shufflevector(lo, hi, 0,1,2,3,4,5,6,7,8,9,10,11,12,13,14,15);
}

// A-fragment (16x32 f16) from row-major staged tile, per ISA 7.12.2:
// lanes 0-15: M=lane, K in {b..b+7} U {b+16..b+23} with b=0; lanes 16-31: b=8.
__device__ __forceinline__ v16h load_afrag(const _Float16* stg, int lane, int kc) {
  int M  = lane & 15;
  int b0 = ((lane < 16) ? 0 : 8) + kc * 32;
  int b1 = b0 + 16;
  v8h lo = *(const v8h*)&stg[M * STRIDE + b0];
  v8h hi = *(const v8h*)&stg[M * STRIDE + b1];
  return cat8(lo, hi);
}

// Weight fragment from the LDS pool: [frag][half][lane][8 halves].
__device__ __forceinline__ v16h load_wfrag(const _Float16* wl, int fi, int lane) {
  v8h lo = *(const v8h*)&wl[((fi * 2 + 0) * 32 + lane) * 8];
  v8h hi = *(const v8h*)&wl[((fi * 2 + 1) * 32 + lane) * 8];
  return cat8(lo, hi);
}

// ---------------------------------------------------------------------------
// Kernel 1: A'[b][q][f] = feats(b,q) . Wg0[f, 0:65] + bg0[f]
//           B [b][p][f] = feats(b,p) . Wg0[f, 65:130]
// feats(b,l) = [x_img[b, :, l], (float)l]
// ---------------------------------------------------------------------------
__global__ void relnet_pre(const float* __restrict__ x,
                           const float* __restrict__ Wg0,
                           const float* __restrict__ bg0,
                           float* __restrict__ Ap, float* __restrict__ Bm) {
  int t = blockIdx.x * blockDim.x + threadIdx.x;      // 16*256*64 threads
  int f = t & 63, q = (t >> 6) & 255, b = t >> 14;
  const float* w  = Wg0 + f * 130;
  const float* xb = x + (size_t)b * 64 * 256;
  float sA = bg0[f], sB = 0.f;
#pragma unroll 8
  for (int ch = 0; ch < 64; ++ch) {
    float v = xb[ch * 256 + q];
    sA += v * w[ch];
    sB += v * w[65 + ch];
  }
  float coord = (float)q;
  sA += coord * w[64];
  sB += coord * w[129];
  int o = (b * 256 + q) * 64 + f;
  Ap[o] = sA;
  Bm[o] = sB;
}

// ---------------------------------------------------------------------------
// Kernel 2: per (b,p) block, loop q in 16-wide tiles:
//   h0 = relu(A'[q] + B[p])           (VALU, staged to LDS as f16)
//   h1 = relu(h0 @ Wg1^T + bg1)       (2 WMMA per 16-col tile, bias in C)
//   h2 = relu(h1 @ Wg2^T + bg2)
//   acc += h2 ; block partial -> Pb[b][p][64]   (deterministic, no atomics)
// ---------------------------------------------------------------------------
__global__ __launch_bounds__(128) void relnet_main(
    const float* __restrict__ Ap, const float* __restrict__ Bm,
    const float* __restrict__ Wg1, const float* __restrict__ bg1,
    const float* __restrict__ Wg2, const float* __restrict__ bg2,
    float* __restrict__ Pb) {
  __shared__ __align__(16) _Float16 st0[4][16 * STRIDE];  // h0 stage (per wave)
  __shared__ __align__(16) _Float16 st1[4][16 * STRIDE];  // h1 stage (per wave)
  // Weight fragment pool: 16 frags (layer,nt,kc) x 2 halves x 32 lanes x 8 f16.
  __shared__ __align__(16) _Float16 wlds[16 * 2 * 32 * 8];
  __shared__ float sBp[64];
  __shared__ float sPart[4][64];

  const int tid  = threadIdx.x;
  const int lane = tid & 31;
  const int wave = tid >> 5;
  const int p = blockIdx.x, b = blockIdx.y;

  if (tid < 64) sBp[tid] = Bm[(size_t)(b * 256 + p) * 64 + tid];

  // Cooperatively build weight fragments, ISA B-layout (32x16 f16):
  // element (K,N) = W[n][k]; lane%16 = N, lane/16 = h, elems = K {8h..8h+7} U {16+8h..16+8h+7}.
  // frag id fi = layer*8 + nt*2 + kc.
#pragma unroll
  for (int e = 0; e < 4; ++e) {
    int idx = tid + 128 * e;            // 512 (frag,lane) entries
    int fi = idx >> 5, li = idx & 31;
    int layer = fi >> 3, nt = (fi >> 1) & 3, kc = fi & 1;
    const float* W = layer ? Wg2 : Wg1;
    int K = kc * 32 + (li >> 4) * 8;
    v8h lo, hi;
#pragma unroll
    for (int j = 0; j < 8; ++j) {
      lo[j] = (_Float16)W[(nt * 16 + (li & 15)) * 64 + K + j];
      hi[j] = (_Float16)W[(nt * 16 + (li & 15)) * 64 + K + 16 + j];
    }
    *(v8h*)&wlds[((fi * 2 + 0) * 32 + li) * 8] = lo;
    *(v8h*)&wlds[((fi * 2 + 1) * 32 + li) * 8] = hi;
  }
  __syncthreads();

  float cb1[4], cb2[4];
#pragma unroll
  for (int nt = 0; nt < 4; ++nt) {
    cb1[nt] = bg1[nt * 16 + (lane & 15)];
    cb2[nt] = bg2[nt * 16 + (lane & 15)];
  }

  v8f acc[4];
#pragma unroll
  for (int nt = 0; nt < 4; ++nt)
#pragma unroll
    for (int r = 0; r < 8; ++r) acc[nt][r] = 0.f;

  _Float16* stg0 = st0[wave];
  _Float16* stg1 = st1[wave];
  const int row = lane & 15;         // pair row this lane stages
  const int kb  = (lane >> 4) * 32;  // feature half this lane stages
  const int mb  = (lane >> 4) * 8;   // D-tile row base for this lane

  for (int t = wave; t < 16; t += 4) {
    const int q0 = t * 16;

    // ---- stage h0 = relu(A'[q] + B[p]) as f16, row-major [16][64] ----
    const v4f* pa = (const v4f*)(Ap + (size_t)((b * 256 + q0 + row) * 64 + kb));
    v4f av[8];
#pragma unroll
    for (int i = 0; i < 8; ++i) av[i] = pa[i];
#pragma unroll
    for (int i = 0; i < 4; ++i) {
      v8h hv;
#pragma unroll
      for (int j = 0; j < 8; ++j) {
        int k = i * 8 + j;
        float s = av[k >> 2][k & 3] + sBp[kb + k];
        hv[j] = (_Float16)fmaxf(s, 0.f);
      }
      *(v8h*)&stg0[row * STRIDE + kb + i * 8] = hv;
    }
    lds_fence();

    v16h a0 = load_afrag(stg0, lane, 0);
    v16h a1 = load_afrag(stg0, lane, 1);

    // ---- layer 1: h1 = relu(h0 @ Wg1^T + bg1), stage transposed to LDS ----
#pragma unroll
    for (int nt = 0; nt < 4; ++nt) {
      v16h wk0 = load_wfrag(wlds, nt * 2 + 0, lane);
      v16h wk1 = load_wfrag(wlds, nt * 2 + 1, lane);
      v8f c;
#pragma unroll
      for (int r = 0; r < 8; ++r) c[r] = cb1[nt];
      v8f d = __builtin_amdgcn_wmma_f32_16x16x32_f16(false, a0, false, wk0,
                                                     (short)0, c, false, false);
      d = __builtin_amdgcn_wmma_f32_16x16x32_f16(false, a1, false, wk1,
                                                 (short)0, d, false, false);
      int N = nt * 16 + (lane & 15);
#pragma unroll
      for (int r = 0; r < 8; ++r)
        stg1[(mb + r) * STRIDE + N] = (_Float16)fmaxf(d[r], 0.f);
    }
    lds_fence();

    v16h g0 = load_afrag(stg1, lane, 0);
    v16h g1 = load_afrag(stg1, lane, 1);

    // ---- layer 2: acc += relu(h1 @ Wg2^T + bg2) ----
#pragma unroll
    for (int nt = 0; nt < 4; ++nt) {
      v16h wk0 = load_wfrag(wlds, 8 + nt * 2 + 0, lane);
      v16h wk1 = load_wfrag(wlds, 8 + nt * 2 + 1, lane);
      v8f c;
#pragma unroll
      for (int r = 0; r < 8; ++r) c[r] = cb2[nt];
      v8f d = __builtin_amdgcn_wmma_f32_16x16x32_f16(false, g0, false, wk0,
                                                     (short)0, c, false, false);
      d = __builtin_amdgcn_wmma_f32_16x16x32_f16(false, g1, false, wk1,
                                                 (short)0, d, false, false);
#pragma unroll
      for (int r = 0; r < 8; ++r) acc[nt][r] += fmaxf(d[r], 0.f);
    }
  }

  // ---- deterministic reduction: sum over pair rows (M) and waves ----
#pragma unroll
  for (int nt = 0; nt < 4; ++nt) {
    float s = 0.f;
#pragma unroll
    for (int r = 0; r < 8; ++r) s += acc[nt][r];
    s += __shfl_xor(s, 16);                 // lanes l and l+16 hold same N
    if (lane < 16) sPart[wave][nt * 16 + lane] = s;
  }
  __syncthreads();
  if (tid < 64) {
    float s = sPart[0][tid] + sPart[1][tid] + sPart[2][tid] + sPart[3][tid];
    Pb[(size_t)(b * 256 + p) * 64 + tid] = s;
  }
}

// ---------------------------------------------------------------------------
// Kernel 3: reduce the 256 per-p partials (fixed order -> deterministic),
// then the tiny head MLP: out = relu(S @ Wp^T + bp) @ Wo^T + bo
// ---------------------------------------------------------------------------
__global__ __launch_bounds__(64) void relnet_head(
    const float* __restrict__ Pb,
    const float* __restrict__ Wp, const float* __restrict__ bp,
    const float* __restrict__ Wo, const float* __restrict__ bo,
    float* __restrict__ out) {
  __shared__ float sS[64], sh[64];
  const int t = threadIdx.x, b = blockIdx.x;
  float s = 0.f;
  for (int p = 0; p < 256; ++p) s += Pb[(size_t)(b * 256 + p) * 64 + t];
  sS[t] = s;
  __syncthreads();
  float a = bp[t];
#pragma unroll 8
  for (int f = 0; f < 64; ++f) a += sS[f] * Wp[t * 64 + f];
  sh[t] = fmaxf(a, 0.f);
  __syncthreads();
  float o = bo[t];
#pragma unroll 8
  for (int f = 0; f < 64; ++f) o += sh[f] * Wo[t * 64 + f];
  out[b * 64 + t] = o;
}

// ---------------------------------------------------------------------------
extern "C" void kernel_launch(void* const* d_in, const int* in_sizes, int n_in,
                              void* d_out, int out_size, void* d_ws, size_t ws_size,
                              hipStream_t stream) {
  const float* x   = (const float*)d_in[0];
  const float* Wg0 = (const float*)d_in[1];
  const float* bg0 = (const float*)d_in[2];
  const float* Wg1 = (const float*)d_in[3];
  const float* bg1 = (const float*)d_in[4];
  const float* Wg2 = (const float*)d_in[5];
  const float* bg2 = (const float*)d_in[6];
  const float* Wp  = (const float*)d_in[7];
  const float* bp  = (const float*)d_in[8];
  const float* Wo  = (const float*)d_in[9];
  const float* bo  = (const float*)d_in[10];
  float* out = (float*)d_out;

  const int BLF = 16 * 256 * 64;        // 262144 floats per plane
  float* Ap = (float*)d_ws;             // [16][256][64] A' (bg0 folded)
  float* Bm = Ap + BLF;                 // [16][256][64] B
  float* Pb = Bm + BLF;                 // [16][256][64] per-block partials

  relnet_pre<<<BLF / 256, 256, 0, stream>>>(x, Wg0, bg0, Ap, Bm);
  relnet_main<<<dim3(256, 16), 128, 0, stream>>>(Ap, Bm, Wg1, bg1, Wg2, bg2, Pb);
  relnet_head<<<16, 64, 0, stream>>>(Pb, Wp, bp, Wo, bo, out);
}
